// OneGATCN_87720412053584
// MI455X (gfx1250) — matrix operand order repaired and checked by reading, not compiled, not run on hardware
//
#include <hip/hip_runtime.h>
#include <hip/hip_bf16.h>
#include <math.h>

typedef __attribute__((ext_vector_type(16))) __bf16        v16bf;
typedef __attribute__((ext_vector_type(8)))  float         v8f;
typedef __attribute__((ext_vector_type(4)))  unsigned int  u32x4;
typedef __attribute__((ext_vector_type(4)))  int           i32x4;
typedef __attribute__((ext_vector_type(8)))  int           i32x8;

// ---------------------------------------------------------------------------
// Problem constants (match reference)
// ---------------------------------------------------------------------------
#define F_IN  512
#define HID   1024
#define C1    512
#define C2    128
#define NC    100
#define NEG_SLOPE 0.2f

#define TM  64
#define TN  128
#define TK  32
#define TKP (TK + 8)   // padded LDS row: 80B stride -> conflict-free, 16B aligned

// Use the Tensor Data Mover when the builtin exists (device pass only; the
// host pass parses the plain-HIP fallback).
#if defined(__HIP_DEVICE_COMPILE__) && defined(__has_builtin)
#  if __has_builtin(__builtin_amdgcn_tensor_load_to_lds)
#    define USE_TDM 1
#  endif
#endif
#ifndef USE_TDM
#  define USE_TDM 0
#endif

// ---------------------------------------------------------------------------
// low 32 bits of a generic pointer to LDS == LDS byte offset (ISA 10.2)
// ---------------------------------------------------------------------------
__device__ __forceinline__ unsigned lds_off(const void* p) {
    return (unsigned)(unsigned long long)p;
}

#if USE_TDM
// 2D tile load via TDM, bf16 elements. dims are the *remaining* tensor
// extents from the tile origin so hardware zero-fills OOB rows/cols.
// LDS padding: +4 DWORDs after every 16 DWORDs (one 64B tile row) ->
// 80B row stride in LDS, matching the [TKP] arrays.
__device__ __forceinline__ void tdm_load_2d(unsigned lds_addr, const void* gptr,
                                            unsigned dim0, unsigned dim1,
                                            unsigned stride0,
                                            unsigned tile0, unsigned tile1)
{
    unsigned long long ga = (unsigned long long)gptr;
    u32x4 g0;
    g0[0] = 1u;                                          // count=1, user mode
    g0[1] = lds_addr;                                    // lds_addr
    g0[2] = (unsigned)(ga & 0xFFFFFFFFu);                // global_addr[31:0]
    g0[3] = (unsigned)((ga >> 32) & 0x1FFFFFFu)          // global_addr[56:32]
          | (2u << 30);                                  // type=2 (image)
    i32x8 g1;
    g1[0] = (int)((1u << 16)                             // data_size=1 (2B)
                | (1u << 20)                             // pad_enable
                | (3u << 22)                             // pad_interval: 16 DWORDs
                | (3u << 25));                           // pad_amount: 4 DWORDs
    g1[1] = (int)((dim0 & 0xFFFFu) << 16);               // tensor_dim0[15:0]
    g1[2] = (int)((dim0 >> 16) | ((dim1 & 0xFFFFu) << 16));
    g1[3] = (int)((dim1 >> 16) | (tile0 << 16));         // tile_dim0
    g1[4] = (int)tile1;                                  // tile_dim1 (tile_dim2=0)
    g1[5] = (int)stride0;                                // tensor_dim0_stride[31:0]
    g1[6] = 0;
    g1[7] = 0;
    i32x4 z4 = {0, 0, 0, 0};
#if __clang_major__ >= 23
    i32x8 z8 = {0, 0, 0, 0, 0, 0, 0, 0};
    __builtin_amdgcn_tensor_load_to_lds(g0, g1, z4, z4, z8, 0);
#else
    __builtin_amdgcn_tensor_load_to_lds(g0, g1, z4, z4, 0);
#endif
}
#endif

// ---------------------------------------------------------------------------
// stage one (A,BT) tile pair into LDS buffers
// A  is [M,K] bf16 row-major; BT is [N,K] bf16 row-major (pre-transposed).
// ---------------------------------------------------------------------------
__device__ __forceinline__ void stage_tiles(__bf16 (*As)[TKP], __bf16 (*Bs)[TKP],
                                            const __bf16* __restrict__ A,
                                            const __bf16* __restrict__ BT,
                                            int M, int N, int K,
                                            int m0, int n0, int k0,
                                            int tid, int wave)
{
#if USE_TDM
    if (wave == 0) {
        tdm_load_2d(lds_off(&As[0][0]), A + (size_t)m0 * K + k0,
                    (unsigned)(K - k0), (unsigned)(M - m0), (unsigned)K,
                    TK, TM);
        tdm_load_2d(lds_off(&Bs[0][0]), BT + (size_t)n0 * K + k0,
                    (unsigned)(K - k0), (unsigned)(N - n0), (unsigned)K,
                    TK, TN);
    }
#else
    // A: 64 rows x 32 cols, 16B per thread, coalesced
    {
        int r = tid >> 2, seg = tid & 3;
        int gr = m0 + r;
        uint4 v = {0u, 0u, 0u, 0u};
        if (gr < M) v = *(const uint4*)(A + (size_t)gr * K + k0 + seg * 8);
        *(uint4*)&As[r][seg * 8] = v;
    }
    // BT: 128 rows x 32 cols, 32B per thread, coalesced
    {
        int n = tid >> 1, seg = tid & 1;
        int gn = n0 + n;
        uint4 v0 = {0u, 0u, 0u, 0u}, v1 = {0u, 0u, 0u, 0u};
        if (gn < N) {
            const __bf16* src = BT + (size_t)gn * K + k0 + seg * 16;
            v0 = *(const uint4*)(src);
            v1 = *(const uint4*)(src + 8);
        }
        *(uint4*)&Bs[n][seg * 16]     = v0;
        *(uint4*)&Bs[n][seg * 16 + 8] = v1;
    }
#endif
}

__device__ __forceinline__ void tdm_wait_pipe() {
#if USE_TDM
    __builtin_amdgcn_s_wait_tensorcnt((short)2);   // next tile's 2 loads in flight
#endif
}
__device__ __forceinline__ void tdm_wait_all() {
#if USE_TDM
    __builtin_amdgcn_s_wait_tensorcnt((short)0);
#endif
}

// ---------------------------------------------------------------------------
// per-wave 32x32 compute step: 2x2 v_wmma_f32_16x16x32_bf16
// A/B element s of lane half h holds k = 16h + s (contiguous k order, not the ISA table); both A and B per-lane data are
// contiguous 8/16-element runs -> ds_load_b128 pairs.
// ---------------------------------------------------------------------------
__device__ __forceinline__ void compute_tile(const __bf16 (*As)[TKP],
                                             const __bf16 (*Bs)[TKP],
                                             v8f acc[2][2],
                                             int wm, int wn, int half, int l16)
{
    v16bf afr[2], bfr[2];
    #pragma unroll
    for (int i = 0; i < 2; ++i) {
        int arow = wm * 32 + i * 16 + l16;
        #pragma unroll
        for (int p = 0; p < 8; ++p) {
            int kk = 2 * p + 16 * half;
            afr[i][2 * p]     = As[arow][kk];
            afr[i][2 * p + 1] = As[arow][kk + 1];
        }
    }
    #pragma unroll
    for (int j = 0; j < 2; ++j) {
        int bcol = wn * 32 + j * 16 + l16;
        #pragma unroll
        for (int p = 0; p < 8; ++p) {
            int kk = 2 * p + 16 * half;
            bfr[j][2 * p]     = Bs[bcol][kk];
            bfr[j][2 * p + 1] = Bs[bcol][kk + 1];
        }
    }
    #pragma unroll
    for (int i = 0; i < 2; ++i)
        #pragma unroll
        for (int j = 0; j < 2; ++j)
            acc[i][j] = __builtin_amdgcn_wmma_f32_16x16x32_bf16(
                false, afr[i], false, bfr[j], (short)0, acc[i][j], false, false);
}

// ---------------------------------------------------------------------------
// bf16 WMMA GEMM: C[M,N] = act(A[M,K] @ BT[N,K]^T + bias)
// TDM double-buffered LDS staging. K % 32 == 0.
// mode: 0 = f32 out, 1 = f32+relu out, 2 = bf16+relu out
// ---------------------------------------------------------------------------
__global__ __launch_bounds__(256)
void gemm_bf16_wmma(const __bf16* __restrict__ A, const __bf16* __restrict__ BT,
                    const float* __restrict__ bias, void* __restrict__ Cout,
                    int M, int N, int K, int mode)
{
    __shared__ __bf16 As[2][TM][TKP];
    __shared__ __bf16 Bs[2][TN][TKP];

    const int tid  = threadIdx.x;
    const int lane = tid & 31;
    const int wave = tid >> 5;
    const int wm   = wave >> 2;   // 0..1
    const int wn   = wave & 3;    // 0..3
    const int half = lane >> 4;
    const int l16  = lane & 15;
    const int m0   = blockIdx.x * TM;
    const int n0   = blockIdx.y * TN;
    const int nk   = K / TK;

    v8f acc[2][2];
    #pragma unroll
    for (int i = 0; i < 2; ++i)
        #pragma unroll
        for (int j = 0; j < 2; ++j) {
            v8f z = {0.f, 0.f, 0.f, 0.f, 0.f, 0.f, 0.f, 0.f};
            acc[i][j] = z;
        }

    stage_tiles(As[0], Bs[0], A, BT, M, N, K, m0, n0, 0, tid, wave);
#if !USE_TDM
    __syncthreads();
#endif

    for (int kt = 0; kt + 1 < nk; ++kt) {
        int nb = (kt + 1) & 1;
        stage_tiles(As[nb], Bs[nb], A, BT, M, N, K, m0, n0, (kt + 1) * TK, tid, wave);
        if (wave == 0) tdm_wait_pipe();
        __syncthreads();
        compute_tile(As[kt & 1], Bs[kt & 1], acc, wm, wn, half, l16);
        __syncthreads();
    }
    if (wave == 0) tdm_wait_all();
    __syncthreads();
    compute_tile(As[(nk - 1) & 1], Bs[(nk - 1) & 1], acc, wm, wn, half, l16);

    // epilogue
    float*  Cf = (float*)Cout;
    __bf16* Ch = (__bf16*)Cout;
    #pragma unroll
    for (int j = 0; j < 2; ++j) {
        int ncol = n0 + wn * 32 + j * 16 + l16;
        if (ncol >= N) continue;
        float bv = bias ? bias[ncol] : 0.0f;
        #pragma unroll
        for (int i = 0; i < 2; ++i) {
            #pragma unroll
            for (int r = 0; r < 8; ++r) {
                int mrow = m0 + wm * 32 + i * 16 + r + 8 * half;
                if (mrow < M) {
                    float v = acc[i][j][r] + bv;
                    if (mode >= 1) v = v > 0.f ? v : 0.f;
                    size_t idx = (size_t)mrow * N + ncol;
                    if (mode == 2) Ch[idx] = (__bf16)v; else Cf[idx] = v;
                }
            }
        }
    }
}

// ---------------------------------------------------------------------------
// f32 -> bf16 conversion, identity layout (n % 4 == 0 for all users)
// ---------------------------------------------------------------------------
__global__ void f32_to_bf16(const float* __restrict__ in, __bf16* __restrict__ out,
                            long long n)
{
    long long i = ((long long)blockIdx.x * blockDim.x + threadIdx.x) * 4;
    if (i + 3 < n) {
        float4 f = *(const float4*)(in + i);
        out[i]     = (__bf16)f.x;
        out[i + 1] = (__bf16)f.y;
        out[i + 2] = (__bf16)f.z;
        out[i + 3] = (__bf16)f.w;
    }
}

// ---------------------------------------------------------------------------
// f32 [K,N] -> bf16 [N,K] transposed conversion (LDS-tiled 32x32)
// ---------------------------------------------------------------------------
__global__ __launch_bounds__(256)
void f32_to_bf16_T(const float* __restrict__ in, __bf16* __restrict__ out,
                   int K, int N)
{
    __shared__ __bf16 tile[32][33];
    int bn = blockIdx.x * 32;
    int bk = blockIdx.y * 32;
    int tx = threadIdx.x & 31;
    int ty = threadIdx.x >> 5;
    #pragma unroll
    for (int r = ty; r < 32; r += 8) {
        int gk = bk + r, gn = bn + tx;
        tile[r][tx] = (gk < K && gn < N) ? (__bf16)in[(size_t)gk * N + gn]
                                         : (__bf16)0.0f;
    }
    __syncthreads();
    #pragma unroll
    for (int r = ty; r < 32; r += 8) {
        int gn = bn + r, gk = bk + tx;
        if (gn < N && gk < K) out[(size_t)gn * K + gk] = tile[tx][r];
    }
}

// ---------------------------------------------------------------------------
// float atomic max via int/uint ordered-encoding trick
// ---------------------------------------------------------------------------
__device__ __forceinline__ void atomicMaxFloat(float* addr, float val) {
    if (val >= 0.0f) atomicMax((int*)addr, __float_as_int(val));
    else             atomicMin((unsigned int*)addr, __float_as_uint(val));
}

__global__ void init_ms(float* __restrict__ m, float* __restrict__ s, int n) {
    int t = blockIdx.x * blockDim.x + threadIdx.x;
    if (t < n) { m[t] = __int_as_float(0xFF800000); s[t] = 0.0f; }
}

__global__ void zero_buf(float* __restrict__ p, long long n) {
    long long t = (long long)blockIdx.x * blockDim.x + threadIdx.x;
    if (t < n) p[t] = 0.0f;
}

// ---------------------------------------------------------------------------
// Pass A: e = att . leaky_relu(xl[src] + xr[dst]); atomic segment-max into m
// ---------------------------------------------------------------------------
__global__ __launch_bounds__(256)
void edge_scores(const float* __restrict__ xl, const float* __restrict__ xr,
                 const float* __restrict__ att, const int* __restrict__ ei,
                 int E, int Nn, float* __restrict__ e_out, float* __restrict__ m)
{
    __shared__ float satt[HID];
    for (int i = threadIdx.x; i < HID; i += blockDim.x) satt[i] = att[i];
    __syncthreads();

    int widx = (int)(((long long)blockIdx.x * blockDim.x + threadIdx.x) >> 5);
    int lane = threadIdx.x & 31;
    int tot  = E + Nn;
    if (widx >= tot) return;

    int src, dst;
    if (widx < E) { src = ei[widx]; dst = ei[E + widx]; }
    else          { src = dst = widx - E; }

    const float* pl = xl + (size_t)src * HID;
    const float* pr = xr + (size_t)dst * HID;

    float acc = 0.0f;
    for (int c = lane; c < HID; c += 32) {
        __builtin_prefetch(pl + c + 256, 0, 1);
        float v = pl[c] + pr[c];
        v = v > 0.0f ? v : NEG_SLOPE * v;
        acc += v * satt[c];
    }
    #pragma unroll
    for (int off = 16; off > 0; off >>= 1)
        acc += __shfl_xor(acc, off, 32);

    if (lane == 0) {
        e_out[widx] = acc;
        atomicMaxFloat(&m[dst], acc);
    }
}

// ---------------------------------------------------------------------------
// Pass B: w = exp(e - m[dst]); atomic segment-sum into s
// ---------------------------------------------------------------------------
__global__ void edge_exp(const float* __restrict__ e, const float* __restrict__ m,
                         const int* __restrict__ ei, int E, int Nn,
                         float* __restrict__ w, float* __restrict__ s)
{
    int t = blockIdx.x * blockDim.x + threadIdx.x;
    if (t >= E + Nn) return;
    int dst = (t < E) ? ei[E + t] : (t - E);
    float val = __expf(e[t] - m[dst]);
    w[t] = val;
    atomicAdd(&s[dst], val);
}

// ---------------------------------------------------------------------------
// Pass C: out[dst] += (w/s[dst]) * xl[src]
// ---------------------------------------------------------------------------
__global__ __launch_bounds__(256)
void edge_scatter(const float* __restrict__ xl, const float* __restrict__ w,
                  const float* __restrict__ s, const int* __restrict__ ei,
                  int E, int Nn, float* __restrict__ out)
{
    int widx = (int)(((long long)blockIdx.x * blockDim.x + threadIdx.x) >> 5);
    int lane = threadIdx.x & 31;
    int tot  = E + Nn;
    if (widx >= tot) return;

    int src, dst;
    if (widx < E) { src = ei[widx]; dst = ei[E + widx]; }
    else          { src = dst = widx - E; }

    float alpha = w[widx] / s[dst];
    const float* pl = xl + (size_t)src * HID;
    float*       po = out + (size_t)dst * HID;
    for (int c = lane; c < HID; c += 32)
        atomicAdd(&po[c], alpha * pl[c]);
}

// ---------------------------------------------------------------------------
// hbf = bf16(relu(acc + conv_b))
// ---------------------------------------------------------------------------
__global__ void bias_relu_bf16(const float* __restrict__ h, const float* __restrict__ b,
                               __bf16* __restrict__ out, long long total)
{
    long long t = (long long)blockIdx.x * blockDim.x + threadIdx.x;
    if (t >= total) return;
    int c = (int)(t & (HID - 1));
    float v = h[t] + b[c];
    out[t] = (__bf16)(v > 0.f ? v : 0.f);
}

// ---------------------------------------------------------------------------
// Launch
// ---------------------------------------------------------------------------
extern "C" void kernel_launch(void* const* d_in, const int* in_sizes, int n_in,
                              void* d_out, int out_size, void* d_ws, size_t ws_size,
                              hipStream_t stream)
{
    const float* x      = (const float*)d_in[0];
    const int*   ei     = (const int*)  d_in[1];
    const float* Wl     = (const float*)d_in[2];
    const float* bl     = (const float*)d_in[3];
    const float* Wr     = (const float*)d_in[4];
    const float* br     = (const float*)d_in[5];
    const float* att    = (const float*)d_in[6];
    const float* conv_b = (const float*)d_in[7];
    const float* W1     = (const float*)d_in[8];
    const float* b1     = (const float*)d_in[9];
    const float* W2     = (const float*)d_in[10];
    const float* b2     = (const float*)d_in[11];
    const float* Wc     = (const float*)d_in[12];
    const float* bc     = (const float*)d_in[13];
    float*       outp   = (float*)d_out;

    const int Nn = in_sizes[0] / F_IN;   // 10000
    const int E  = in_sizes[1] / 2;      // 160000
    const int ET = E + Nn;

    // ---- workspace carve-out ----
    float* xl = (float*)d_ws;                      // [Nn,HID] f32; later: hbf (bf16)
    float* xb = xl + (size_t)Nn * HID;             // [Nn,HID] f32: xr -> accum; later: h2bf
    float* ev = xb + (size_t)Nn * HID;             // [ET]
    float* wv = ev + ET;                           // [ET]
    float* mv = wv + ET;                           // [Nn]
    float* sv = mv + Nn;                           // [Nn]
    __bf16* xbf  = (__bf16*)(sv + Nn);             // [Nn,F_IN]; later: h1bf [Nn,C1]
    __bf16* wlT  = xbf + (size_t)Nn * F_IN;        // [HID,F_IN]
    __bf16* wrT  = wlT + (size_t)HID * F_IN;       // [HID,F_IN]
    __bf16* w1T  = wrT + (size_t)HID * F_IN;       // [C1,HID]
    __bf16* w2T  = w1T + (size_t)C1 * HID;         // [C2,C1]
    __bf16* wcT  = w2T + (size_t)C2 * C1;          // [NC,C2]
    __bf16* hbf  = (__bf16*)xl;                    // [Nn,HID]  (xl consumed by then)
    __bf16* h1bf = xbf;                            // [Nn,C1]   (x consumed by then)
    __bf16* h2bf = (__bf16*)xb;                    // [Nn,C2]   (accum consumed by then)

    dim3 blk(256);
    auto cdiv = [](long long a, long long b) { return (unsigned)((a + b - 1) / b); };

    // 0) init segment max/sum
    init_ms<<<dim3(cdiv(Nn, 256)), blk, 0, stream>>>(mv, sv, Nn);

    // 1) convert activations (identity) and weights (transposed) to bf16
    f32_to_bf16<<<dim3(cdiv((long long)Nn * F_IN / 4, 256)), blk, 0, stream>>>(
        x, xbf, (long long)Nn * F_IN);
    f32_to_bf16_T<<<dim3(cdiv(HID, 32), cdiv(F_IN, 32)), blk, 0, stream>>>(Wl, wlT, F_IN, HID);
    f32_to_bf16_T<<<dim3(cdiv(HID, 32), cdiv(F_IN, 32)), blk, 0, stream>>>(Wr, wrT, F_IN, HID);
    f32_to_bf16_T<<<dim3(cdiv(C1, 32),  cdiv(HID, 32)),  blk, 0, stream>>>(W1, w1T, HID, C1);
    f32_to_bf16_T<<<dim3(cdiv(C2, 32),  cdiv(C1, 32)),   blk, 0, stream>>>(W2, w2T, C1, C2);
    f32_to_bf16_T<<<dim3(cdiv(NC, 32),  cdiv(C2, 32)),   blk, 0, stream>>>(Wc, wcT, C2, NC);

    // 2) xl = x @ Wl + bl ; xr = x @ Wr + br   (f32 out)
    {
        dim3 grid((Nn + TM - 1) / TM, (HID + TN - 1) / TN);
        gemm_bf16_wmma<<<grid, blk, 0, stream>>>(xbf, wlT, bl, xl, Nn, HID, F_IN, 0);
        gemm_bf16_wmma<<<grid, blk, 0, stream>>>(xbf, wrT, br, xb, Nn, HID, F_IN, 0);
    }

    // 3) per-edge attention logits + segment max
    edge_scores<<<dim3(cdiv((long long)ET * 32, 256)), blk, 0, stream>>>(
        xl, xb, att, ei, E, Nn, ev, mv);

    // 4) xr consumed -> reuse xb as output accumulator
    zero_buf<<<dim3(cdiv((long long)Nn * HID, 256)), blk, 0, stream>>>(xb, (long long)Nn * HID);

    // 5) w = exp(e - m[dst]); segment sum s
    edge_exp<<<dim3(cdiv(ET, 256)), blk, 0, stream>>>(ev, mv, ei, E, Nn, wv, sv);

    // 6) out[dst] += alpha * xl[src]
    edge_scatter<<<dim3(cdiv((long long)ET * 32, 256)), blk, 0, stream>>>(
        xl, wv, sv, ei, E, Nn, xb);

    // 7) hbf = bf16(relu(accum + conv_b))   (xl region reused)
    bias_relu_bf16<<<dim3(cdiv((long long)Nn * HID, 256)), blk, 0, stream>>>(
        xb, conv_b, hbf, (long long)Nn * HID);

    // 8) h1 = relu(h @ W1 + b1) -> bf16
    {
        dim3 grid((Nn + TM - 1) / TM, (C1 + TN - 1) / TN);
        gemm_bf16_wmma<<<grid, blk, 0, stream>>>(hbf, w1T, b1, h1bf, Nn, C1, HID, 2);
    }
    // 9) h2 = relu(h1 @ W2 + b2) -> bf16
    {
        dim3 grid((Nn + TM - 1) / TM, (C2 + TN - 1) / TN);
        gemm_bf16_wmma<<<grid, blk, 0, stream>>>(h1bf, w2T, b2, h2bf, Nn, C2, C1, 2);
    }
    // 10) logits = h2 @ Wc + bc -> d_out (f32)
    {
        dim3 grid((Nn + TM - 1) / TM, (NC + TN - 1) / TN);
        gemm_bf16_wmma<<<grid, blk, 0, stream>>>(h2bf, wcT, bc, outp, Nn, NC, C2, 0);
    }
}
